// MutiHeadAttention_6210522710067
// MI455X (gfx1250) — matrix operand order repaired and checked by reading, not compiled, not run on hardware
//
#include <hip/hip_runtime.h>

// ---------------------------------------------------------------------------
// Multi-head attention for MI455X (gfx1250): bf16 WMMA + f32 accumulate,
// double-buffered TDM staging, DPP row reductions, WMMA row-sums.
// ---------------------------------------------------------------------------

typedef __bf16 bf16;
typedef __attribute__((ext_vector_type(16))) __bf16 v16bf;
typedef __attribute__((ext_vector_type(8)))  float  v8f;
typedef __attribute__((ext_vector_type(4)))  unsigned int u32x4;
typedef __attribute__((ext_vector_type(8)))  int i32x8;
typedef __attribute__((ext_vector_type(4)))  int i32x4;

#define D_MODEL 1024
#define HEADS   16
#define DK      64
#define SEQ     2048
#define BATCH   2
#define BS      (BATCH * SEQ)   // 4096 rows
#define NKB     (SEQ / 32)      // key blocks per head

static __device__ __forceinline__ v8f wmma_bf16(v16bf a, v16bf b, v8f c) {
    return __builtin_amdgcn_wmma_f32_16x16x32_bf16(false, a, false, b,
                                                   (short)0, c, false, false);
}

union Frag16 { uint4 u[2]; v16bf v; };
union Pack4  { bf16 h[4]; uint2 u2; };
union Pack8  { bf16 h[8]; uint4 u4; };

// Low 32 bits of a flat shared-aperture address == LDS byte offset.
static __device__ __forceinline__ unsigned lds_offset(const void* p) {
    return (unsigned)(uintptr_t)p;
}

// ---- Tensor Data Mover: 2-D bf16 tile load, global -> LDS (D# per ISA §8) --
static __device__ __forceinline__ void tdm_load_2d_bf16(const void* gaddr,
                                                        unsigned lds_off,
                                                        unsigned tile_d0,
                                                        unsigned tile_d1,
                                                        unsigned tensor_d0,
                                                        unsigned tensor_d1,
                                                        unsigned stride0) {
    u32x4 g0;
    g0[0] = 1u;                                   // count=1, user mode
    g0[1] = lds_off;                              // lds_addr
    unsigned long long ga = (unsigned long long)(uintptr_t)gaddr;
    g0[2] = (unsigned)ga;                         // global_addr[31:0]
    g0[3] = (unsigned)(ga >> 32) | (2u << 30);    // global_addr[56:32] | type=2

    i32x8 g1;
    g1[0] = (int)(1u << 16);                      // data_size=1 (2 bytes)
    g1[1] = (int)(tensor_d0 << 16);               // tensor_dim0[15:0]
    g1[2] = (int)((tensor_d0 >> 16) | (tensor_d1 << 16));
    g1[3] = (int)((tensor_d1 >> 16) | (tile_d0 << 16));
    g1[4] = (int)(tile_d1 & 0xFFFFu);             // tile_dim1 | tile_dim2=0
    g1[5] = (int)stride0;                         // tensor_dim0_stride[31:0]
    g1[6] = 0;
    g1[7] = 0;

    i32x4 gz4; gz4[0] = gz4[1] = gz4[2] = gz4[3] = 0;
    i32x8 gz8;
#pragma unroll
    for (int i = 0; i < 8; ++i) gz8[i] = 0;
    __builtin_amdgcn_tensor_load_to_lds(g0, g1, gz4, gz4, gz8, 0);
}

// ---- DPP rotate reduction over the 16-lane row (wave32 halves) -------------
template <int ROT>
static __device__ __forceinline__ float row_ror(float x) {
    int r = __builtin_amdgcn_update_dpp(0, __builtin_bit_cast(int, x),
                                        0x120 | ROT, 0xF, 0xF, true);
    return __builtin_bit_cast(float, r);
}
static __device__ __forceinline__ float row_max16(float x) {
    x = fmaxf(x, row_ror<8>(x));
    x = fmaxf(x, row_ror<4>(x));
    x = fmaxf(x, row_ror<2>(x));
    x = fmaxf(x, row_ror<1>(x));
    return x;
}

// ---------------------------------------------------------------------------
// GEMM: Y = (X @ W^T + bias) * scale.
// IN_BF16:  X is bf16 (A tile staged by TDM); else f32, converted on the fly.
// HEAD_OUT: bf16 into [B,H,S,DK]; else f32 row-major.
// 256 thr = 8 waves; block tile 128x128; wave tile 32x64; K-step 64.
// ---------------------------------------------------------------------------
template <bool IN_BF16, bool HEAD_OUT>
__global__ __launch_bounds__(256) void gemm_kernel(const void* __restrict__ Xv,
                                                   const float* __restrict__ W,
                                                   const float* __restrict__ bias,
                                                   void* __restrict__ Yv,
                                                   float scale) {
    __shared__ __align__(16) bf16 As[128][64];
    __shared__ __align__(16) bf16 Bs[128][64];

    const int tid   = threadIdx.x;
    const int lane  = tid & 31;
    const int wid   = tid >> 5;
    const int waveM = wid & 3;
    const int waveN = wid >> 2;
    const int tileN = blockIdx.x * 128;
    const int tileM = blockIdx.y * 128;

    v8f acc[2][4];
#pragma unroll
    for (int mf = 0; mf < 2; ++mf)
#pragma unroll
        for (int nf = 0; nf < 4; ++nf) {
            v8f z = {0.f, 0.f, 0.f, 0.f, 0.f, 0.f, 0.f, 0.f};
            acc[mf][nf] = z;
        }

    const int k0a  = (lane < 16) ? 0 : 8;
    const int k0b  = (lane < 16) ? 0 : 8;
    const int colL = lane & 15;

    for (int kt = 0; kt < D_MODEL / 64; ++kt) {
        __syncthreads();
        const int k0 = kt * 64;

        // ---- stage A tile (128 x 64 bf16) ----
        if constexpr (IN_BF16) {
            if (wid == 0) {
                const bf16* X = (const bf16*)Xv;
                tdm_load_2d_bf16(X + (size_t)tileM * D_MODEL + k0,
                                 lds_offset(&As[0][0]),
                                 /*tile*/ 64, 128,
                                 /*tensor*/ D_MODEL, BS,
                                 /*stride0*/ D_MODEL);
            }
        } else {
            const float* X = (const float*)Xv;
#pragma unroll
            for (int i = 0; i < 8; ++i) {
                int id = tid + i * 256;              // 2048 float4 chunks
                int row = id >> 4, c4 = id & 15;
                const float* src = &X[(size_t)(tileM + row) * D_MODEL + k0 + c4 * 4];
                __builtin_prefetch(src + 64, 0, 0);  // next K tile
                float4 f = *(const float4*)src;
                Pack4 p;
                p.h[0] = (bf16)f.x; p.h[1] = (bf16)f.y;
                p.h[2] = (bf16)f.z; p.h[3] = (bf16)f.w;
                *(uint2*)&As[row][c4 * 4] = p.u2;
            }
        }
        // ---- stage B tile (128 x 64) from W (f32 -> bf16) ----
#pragma unroll
        for (int i = 0; i < 8; ++i) {
            int id = tid + i * 256;
            int row = id >> 4, c4 = id & 15;
            const float* src = &W[(size_t)(tileN + row) * D_MODEL + k0 + c4 * 4];
            __builtin_prefetch(src + 64, 0, 0);
            float4 f = *(const float4*)src;
            Pack4 p;
            p.h[0] = (bf16)f.x; p.h[1] = (bf16)f.y;
            p.h[2] = (bf16)f.z; p.h[3] = (bf16)f.w;
            *(uint2*)&Bs[row][c4 * 4] = p.u2;
        }
        if constexpr (IN_BF16) {
            if (wid == 0) __builtin_amdgcn_s_wait_tensorcnt(0);
        }
        __syncthreads();

        // ---- two K-halves; batch ALL fragment loads before the WMMAs ----
#pragma unroll
        for (int kf = 0; kf < 2; ++kf) {
            const int kb0 = kf * 32;
            Frag16 a[2], b[4];
#pragma unroll
            for (int mf = 0; mf < 2; ++mf) {
                int row = waveM * 32 + mf * 16 + colL;
                a[mf].u[0] = *(const uint4*)&As[row][kb0 + k0a];
                a[mf].u[1] = *(const uint4*)&As[row][kb0 + k0a + 16];
            }
#pragma unroll
            for (int nf = 0; nf < 4; ++nf) {
                int n = waveN * 64 + nf * 16 + colL;
                b[nf].u[0] = *(const uint4*)&Bs[n][kb0 + k0b];
                b[nf].u[1] = *(const uint4*)&Bs[n][kb0 + k0b + 16];
            }
#pragma unroll
            for (int mf = 0; mf < 2; ++mf)
#pragma unroll
                for (int nf = 0; nf < 4; ++nf)
                    acc[mf][nf] = wmma_bf16(a[mf].v, b[nf].v, acc[mf][nf]);
        }
    }

    // ---- epilogue: (acc + bias) * scale ----
    const int rowOff = (lane < 16) ? 0 : 8;
#pragma unroll
    for (int mf = 0; mf < 2; ++mf)
#pragma unroll
        for (int nf = 0; nf < 4; ++nf)
#pragma unroll
            for (int r = 0; r < 8; ++r) {
                int m = tileM + waveM * 32 + mf * 16 + r + rowOff;
                int n = tileN + waveN * 64 + nf * 16 + colL;
                float val = (acc[mf][nf][r] + bias[n]) * scale;
                if constexpr (HEAD_OUT) {
                    bf16* Y = (bf16*)Yv;
                    int bb = m >> 11, ss = m & 2047;
                    int hh = n >> 6,  dk = n & 63;
                    Y[((((size_t)bb * HEADS + hh) * SEQ + ss) << 6) + dk] = (bf16)val;
                } else {
                    float* Y = (float*)Yv;
                    Y[(size_t)m * D_MODEL + n] = val;
                }
            }
}

// ---------------------------------------------------------------------------
// Flash attention over bf16 head tensors [B,H,S,DK].  Q pre-scaled by
// 1/sqrt(DK).  K tiles double-buffered via TDM: the DMA for block kb+1 is
// issued before computing block kb and overlaps the whole iteration.
// V staged transposed by all waves.  Row max via DPP; row sums via P@ones.
// ---------------------------------------------------------------------------
__global__ __launch_bounds__(256) void attention_kernel(const bf16* __restrict__ Qh,
                                                        const bf16* __restrict__ Kh,
                                                        const bf16* __restrict__ Vh,
                                                        bf16* __restrict__ Ctx) {
    __shared__ __align__(16) bf16 Ks[2][32][64];   // [buf][key][dk] (TDM dest)
    __shared__ __align__(16) bf16 Vt[64][32];      // [dk][key]   (transposed)
    __shared__ __align__(16) bf16 Pl[8][16][32];   // per-wave P scratch

    const int tid  = threadIdx.x;
    const int lane = tid & 31;
    const int wid  = tid >> 5;
    const int bh   = blockIdx.y;           // b*16 + h
    const int b    = bh >> 4;
    const int h    = bh & 15;
    const size_t headBase = (size_t)bh * SEQ * DK;

    const int qrow0 = blockIdx.x * 128 + wid * 16;

    const int rowOff = (lane < 16) ? 0 : 8;
    const int colL   = lane & 15;
    const int k0a    = (lane < 16) ? 0 : 8;
    const int k0b    = (lane < 16) ? 0 : 8;

    // ---- Q fragments (resident all loop) ----
    Frag16 qa[2];
    {
        const bf16* qp = Qh + headBase + (size_t)(qrow0 + colL) * DK;
        qa[0].u[0] = *(const uint4*)&qp[k0a];
        qa[0].u[1] = *(const uint4*)&qp[k0a + 16];
        qa[1].u[0] = *(const uint4*)&qp[32 + k0a];
        qa[1].u[1] = *(const uint4*)&qp[32 + k0a + 16];
    }
    // ---- all-ones B fragment (row-sum trick) ----
    Frag16 onesF;
    {
        Pack8 po;
#pragma unroll
        for (int j = 0; j < 8; ++j) po.h[j] = (bf16)1.0f;
        onesF.u[0] = po.u4;
        onesF.u[1] = po.u4;
    }

    v8f cc[4];
#pragma unroll
    for (int df = 0; df < 4; ++df) {
        v8f z = {0.f, 0.f, 0.f, 0.f, 0.f, 0.f, 0.f, 0.f};
        cc[df] = z;
    }
    float mI[8], lI[8];
#pragma unroll
    for (int r = 0; r < 8; ++r) { mI[r] = -1e30f; lI[r] = 0.f; }

    // ---- prologue: kick off DMA of the first K tile ----
    if (wid == 0) {
        tdm_load_2d_bf16(Kh + headBase, lds_offset(&Ks[0][0][0]),
                         /*tile*/ DK, 32, /*tensor*/ DK, SEQ, /*stride0*/ DK);
    }

    for (int kb = 0; kb < NKB; ++kb) {
        __syncthreads();   // all waves done reading Ks[(kb+1)&1] (iter kb-1) and Vt

        // ---- issue DMA for the NEXT K tile (overlaps this iteration) ----
        if (kb + 1 < NKB) {
            if (wid == 0) {
                tdm_load_2d_bf16(Kh + headBase + (size_t)(kb + 1) * 32 * DK,
                                 lds_offset(&Ks[(kb + 1) & 1][0][0]),
                                 /*tile*/ DK, 32, /*tensor*/ DK, SEQ, /*stride0*/ DK);
            }
        }
        // ---- stage V transposed (overlaps the in-flight DMA) ----
        {
            int key = tid >> 3;
            int dk0 = (tid & 7) * 8;
            const size_t g = headBase + (size_t)(kb * 32 + key) * DK + dk0;
            Pack8 pv; pv.u4 = *(const uint4*)&Vh[g];
#pragma unroll
            for (int j = 0; j < 8; ++j) Vt[dk0 + j][key] = pv.h[j];
        }
        // TDM in-order per wave: <=1 outstanding means tile kb has landed.
        if (wid == 0) {
            if (kb + 1 < NKB) __builtin_amdgcn_s_wait_tensorcnt(1);
            else              __builtin_amdgcn_s_wait_tensorcnt(0);
        }
        __syncthreads();

        const bf16 (*Kc)[64] = Ks[kb & 1];

        // ---- scores: batch all 4 K fragments, then 4 WMMAs ----
        Frag16 kfr[2][2];
#pragma unroll
        for (int nf = 0; nf < 2; ++nf) {
            int key = nf * 16 + colL;
            kfr[nf][0].u[0] = *(const uint4*)&Kc[key][k0b];
            kfr[nf][0].u[1] = *(const uint4*)&Kc[key][k0b + 16];
            kfr[nf][1].u[0] = *(const uint4*)&Kc[key][32 + k0b];
            kfr[nf][1].u[1] = *(const uint4*)&Kc[key][32 + k0b + 16];
        }
        v8f sf[2];
#pragma unroll
        for (int nf = 0; nf < 2; ++nf) {
            v8f s = {0.f, 0.f, 0.f, 0.f, 0.f, 0.f, 0.f, 0.f};
            s = wmma_bf16(qa[0].v, kfr[nf][0].v, s);
            s = wmma_bf16(qa[1].v, kfr[nf][1].v, s);
            sf[nf] = s;
        }

        // ---- online softmax: DPP row-max, exp, rescale ----
        float mNew[8], alpha[8];
#pragma unroll
        for (int r = 0; r < 8; ++r) {
            float mx = row_max16(fmaxf(sf[0][r], sf[1][r]));
            mNew[r]  = fmaxf(mI[r], mx);
            alpha[r] = __expf(mI[r] - mNew[r]);
            mI[r]    = mNew[r];
        }
#pragma unroll
        for (int nf = 0; nf < 2; ++nf)
#pragma unroll
            for (int r = 0; r < 8; ++r)
                sf[nf][r] = __expf(sf[nf][r] - mNew[r]);
#pragma unroll
        for (int df = 0; df < 4; ++df)
#pragma unroll
            for (int r = 0; r < 8; ++r)
                cc[df][r] *= alpha[r];

        // ---- transpose P through per-wave LDS into A-frag layout ----
#pragma unroll
        for (int nf = 0; nf < 2; ++nf)
#pragma unroll
            for (int r = 0; r < 8; ++r)
                Pl[wid][r + rowOff][nf * 16 + colL] = (bf16)sf[nf][r];
        __syncthreads();

        // ---- batch P + all 4 V fragments, then 5 WMMAs ----
        Frag16 pa, vb[4];
        pa.u[0] = *(const uint4*)&Pl[wid][colL][k0a];
        pa.u[1] = *(const uint4*)&Pl[wid][colL][k0a + 16];
#pragma unroll
        for (int df = 0; df < 4; ++df) {
            int dk = df * 16 + colL;
            vb[df].u[0] = *(const uint4*)&Vt[dk][k0b];
            vb[df].u[1] = *(const uint4*)&Vt[dk][k0b + 16];
        }
        {
            v8f z = {0.f, 0.f, 0.f, 0.f, 0.f, 0.f, 0.f, 0.f};
            v8f srow = wmma_bf16(pa.v, onesF.v, z);   // row sums
#pragma unroll
            for (int r = 0; r < 8; ++r)
                lI[r] = lI[r] * alpha[r] + srow[r];
        }
#pragma unroll
        for (int df = 0; df < 4; ++df)
            cc[df] = wmma_bf16(pa.v, vb[df].v, cc[df]);
    }

    // ---- normalize and store ctx into concat layout [B,S,D_MODEL] bf16 ----
#pragma unroll
    for (int r = 0; r < 8; ++r) {
        float inv = 1.0f / lI[r];
#pragma unroll
        for (int df = 0; df < 4; ++df) cc[df][r] *= inv;
    }
#pragma unroll
    for (int df = 0; df < 4; ++df)
#pragma unroll
        for (int r = 0; r < 8; ++r) {
            int srow = qrow0 + r + rowOff;
            int col  = h * 64 + df * 16 + colL;
            Ctx[((size_t)b * SEQ + srow) * D_MODEL + col] = (bf16)cc[df][r];
        }
}

// ---------------------------------------------------------------------------
extern "C" void kernel_launch(void* const* d_in, const int* in_sizes, int n_in,
                              void* d_out, int out_size, void* d_ws, size_t ws_size,
                              hipStream_t stream) {
    const float* q  = (const float*)d_in[0];
    const float* k  = (const float*)d_in[1];
    const float* v  = (const float*)d_in[2];
    const float* wq = (const float*)d_in[3];
    const float* bq = (const float*)d_in[4];
    const float* wk = (const float*)d_in[5];
    const float* bk = (const float*)d_in[6];
    const float* wv = (const float*)d_in[7];
    const float* bv = (const float*)d_in[8];
    const float* wo = (const float*)d_in[9];
    const float* bo = (const float*)d_in[10];
    float* out = (float*)d_out;

    char* ws = (char*)d_ws;
    const size_t HSZ = (size_t)BS * D_MODEL * sizeof(bf16);  // 8 MB each
    bf16* Qh  = (bf16*)(ws);
    bf16* Kh  = (bf16*)(ws + HSZ);
    bf16* Vh  = (bf16*)(ws + 2 * HSZ);
    bf16* Ctx = (bf16*)(ws + 3 * HSZ);

    dim3 blk(256);
    dim3 gProj(D_MODEL / 128, BS / 128);   // (8, 32)

    const float qscale = 0.125f;           // 1/sqrt(DK), folded into Q proj
    gemm_kernel<false, true><<<gProj, blk, 0, stream>>>((const void*)q, wq, bq, (void*)Qh, qscale);
    gemm_kernel<false, true><<<gProj, blk, 0, stream>>>((const void*)k, wk, bk, (void*)Kh, 1.0f);
    gemm_kernel<false, true><<<gProj, blk, 0, stream>>>((const void*)v, wv, bv, (void*)Vh, 1.0f);

    attention_kernel<<<dim3(SEQ / 128, BATCH * HEADS), blk, 0, stream>>>(Qh, Kh, Vh, Ctx);

    gemm_kernel<true, false><<<gProj, blk, 0, stream>>>((const void*)Ctx, wo, bo, (void*)out, 1.0f);
}
